// GResBlock_86096914415861
// MI455X (gfx1250) — matrix operand order repaired and checked by reading, not compiled, not run on hardware
//
#include <hip/hip_runtime.h>
#include <hip/hip_bf16.h>

// ---------------------------------------------------------------------------
// GCN residual block for MI455X (gfx1250, wave32, WMMA).
//   N_NODES=50000 (=16*3125), N_EDGES=800000, DIM=256.
// GEMMs via v_wmma_f32_16x16x32_f16; gather/scatter stays L2-resident
// (H f16 = 25.6MB, f32 accumulator = 51.2MB, both << 192MB L2).
// ---------------------------------------------------------------------------

typedef __attribute__((ext_vector_type(16))) _Float16 v16h;
typedef __attribute__((ext_vector_type(8)))  float    v8f;

#define N_NODES 50000
#define N_EDGES 800000
#define DIM     256

// ---- workspace layout (bytes, 256B aligned) -------------------------------
#define OFF_DINV  ((size_t)0)                         // 50000 f32
#define OFF_XH    ((size_t)200192)                    // 50000*256 f16 (GEMM input)
#define OFF_HG    (OFF_XH  + (size_t)25600000)        // 50000*256 f16 (GEMM output)
#define OFF_AGG   (OFF_HG  + (size_t)25600000)        // 50000*256 f32 (accumulator)
#define OFF_W1T   (OFF_AGG + (size_t)51200000)        // 256*256 f16
#define OFF_W2T   (OFF_W1T + (size_t)131072)
// total ~102.9 MB

// ---------------------------------------------------------------------------
// degree / normalization
// ---------------------------------------------------------------------------
__global__ void deg_init_kernel(float* __restrict__ deg, int n) {
    int i = blockIdx.x * blockDim.x + threadIdx.x;
    if (i < n) deg[i] = 1.0f;                         // self loop
}

__global__ void deg_count_kernel(const int* __restrict__ dst, float* __restrict__ deg, int ne) {
    int i = blockIdx.x * blockDim.x + threadIdx.x;
    if (i < ne) atomicAdd(&deg[dst[i]], 1.0f);
}

__global__ void dinv_kernel(float* __restrict__ deg, int n) {
    int i = blockIdx.x * blockDim.x + threadIdx.x;
    if (i < n) deg[i] = rsqrtf(deg[i]);               // deg >= 1 always
}

// ---------------------------------------------------------------------------
// conversions
// ---------------------------------------------------------------------------
__global__ void f32_to_f16_kernel(const float* __restrict__ in, _Float16* __restrict__ out, int n) {
    int i = blockIdx.x * blockDim.x + threadIdx.x;
    if (i < n) out[i] = (_Float16)in[i];
}

// Wt[n][k] = (f16) W[k][n]   (256x256)
__global__ void transpose_w_kernel(const float* __restrict__ W, _Float16* __restrict__ Wt) {
    int t = blockIdx.x * blockDim.x + threadIdx.x;    // 65536 threads
    int n = t >> 8;
    int k = t & 255;
    Wt[n * DIM + k] = (_Float16)W[k * DIM + n];
}

// ---------------------------------------------------------------------------
// WMMA fragment loaders (ISA 7.12.2, wave32, 16-bit data)
// ---------------------------------------------------------------------------
// A 16x32 (MxK), row-major source, ld = DIM.
// lanes 0-15: M=lane, K = {0..7, 16..23}; lanes 16-31: M=lane-16, K = {8..15, 24..31}
__device__ __forceinline__ v16h load_frag_a(const _Float16* __restrict__ p, int lane) {
    int row   = lane & 15;
    int kbase = (lane >> 4) << 3;                     // 0 or 8
    const _Float16* r = p + row * DIM;
    v16h f;
#pragma unroll
    for (int h = 0; h < 8; ++h) f[h]     = r[kbase + h];
#pragma unroll
    for (int h = 0; h < 8; ++h) f[8 + h] = r[kbase + 16 + h];
    return f;
}

// B 32x16 (KxN), sourced from Wt (row n = output column), contiguous K per lane.
// lanes 0-15: col=lane, K = {0..7, 16..23}; lanes 16-31: col=lane-16, K = {8..15, 24..31}
__device__ __forceinline__ v16h load_frag_b(const _Float16* __restrict__ p, int lane) {
    int col   = lane & 15;
    int kbase = (lane >> 4) << 3;                     // 0 or 8
    const _Float16* r = p + col * DIM + kbase;
    v16h f;
#pragma unroll
    for (int h = 0; h < 8; ++h) { f[h] = r[h]; f[8 + h] = r[16 + h]; }
    return f;
}

// ---------------------------------------------------------------------------
// GEMM: C[M x 256] = A[M x 256] * Wt^T, f16 in / f32 accum / f16 out.
// One wave per 16x16 output tile; 8 x v_wmma_f32_16x16x32_f16 per wave.
// grid: (M/16)*16 waves = M waves; block = 256 threads = 8 waves.
// ---------------------------------------------------------------------------
__global__ __launch_bounds__(256) void gemm_wmma_kernel(const _Float16* __restrict__ A,
                                                        const _Float16* __restrict__ Bt,
                                                        _Float16* __restrict__ C) {
    int wave = (blockIdx.x * blockDim.x + threadIdx.x) >> 5;
    int lane = threadIdx.x & 31;
    int tm = wave >> 4;                               // 0..3124
    int tn = wave & 15;                               // 0..15
    const _Float16* a0 = A  + (size_t)tm * 16 * DIM;
    const _Float16* b0 = Bt + (size_t)tn * 16 * DIM;

    v8f c = {};
#pragma unroll
    for (int kk = 0; kk < DIM; kk += 32) {
        v16h a = load_frag_a(a0 + kk, lane);
        v16h b = load_frag_b(b0 + kk, lane);
        c = __builtin_amdgcn_wmma_f32_16x16x32_f16(
                /*neg_a=*/false, a, /*neg_b=*/false, b,
                /*c_mod=*/(short)0, c, /*reuse_a=*/false, /*reuse_b=*/false);
    }

    // C layout: N = lane&15, M = r + 8*(lane>=16)
    int n     = lane & 15;
    int mbase = (lane >> 4) << 3;
    _Float16* crow = C + ((size_t)tm * 16 + mbase) * DIM + tn * 16 + n;
#pragma unroll
    for (int r = 0; r < 8; ++r) crow[(size_t)r * DIM] = (_Float16)c[r];
}

// ---------------------------------------------------------------------------
// agg[i][j] = b[j] + dinv[i]^2 * H[i][j]   (self-loop + bias init)
// ---------------------------------------------------------------------------
__global__ void init_agg_kernel(const float* __restrict__ bias,
                                const float* __restrict__ dinv,
                                const _Float16* __restrict__ H,
                                float* __restrict__ agg) {
    int t = blockIdx.x * blockDim.x + threadIdx.x;    // N_NODES*256 threads
    int i = t >> 8;
    int j = t & 255;
    float dv = dinv[i];
    agg[t] = bias[j] + dv * dv * (float)H[t];
}

// ---------------------------------------------------------------------------
// per-edge: agg[dst] += dinv[src]*dinv[dst] * H[src]   (one wave per edge)
// 32 lanes x 8 columns; H rows (512B f16) + agg rows (1KB f32) are L2-resident
// ---------------------------------------------------------------------------
__global__ __launch_bounds__(256) void edge_scatter_kernel(const int* __restrict__ src,
                                                           const int* __restrict__ dst,
                                                           const float* __restrict__ dinv,
                                                           const _Float16* __restrict__ H,
                                                           float* __restrict__ agg) {
    int t = blockIdx.x * blockDim.x + threadIdx.x;
    int e = t >> 5;
    if (e >= N_EDGES) return;
    int lane = t & 31;
    int s = src[e];
    int d = dst[e];
    float norm = dinv[s] * dinv[d];
    const _Float16* hrow = H   + (size_t)s * DIM + lane * 8;
    float*          arow = agg + (size_t)d * DIM + lane * 8;
#pragma unroll
    for (int j = 0; j < 8; ++j)
        atomicAdd(&arow[j], norm * (float)hrow[j]);
}

// ---------------------------------------------------------------------------
// next-layer input: f16(relu(agg))
// ---------------------------------------------------------------------------
__global__ void relu_f16_kernel(const float* __restrict__ agg, _Float16* __restrict__ out, int n) {
    int i = blockIdx.x * blockDim.x + threadIdx.x;
    if (i < n) out[i] = (_Float16)fmaxf(agg[i], 0.0f);
}

// ---------------------------------------------------------------------------
// out = 0.5 * (x + relu(agg))
// ---------------------------------------------------------------------------
__global__ void finalize_kernel(const float* __restrict__ x, const float* __restrict__ agg,
                                float* __restrict__ out, int n) {
    int i = blockIdx.x * blockDim.x + threadIdx.x;
    if (i < n) out[i] = 0.5f * (x[i] + fmaxf(agg[i], 0.0f));
}

// ---------------------------------------------------------------------------
extern "C" void kernel_launch(void* const* d_in, const int* in_sizes, int n_in,
                              void* d_out, int out_size, void* d_ws, size_t ws_size,
                              hipStream_t stream) {
    (void)in_sizes; (void)n_in; (void)out_size; (void)ws_size;

    const float* x   = (const float*)d_in[0];         // [1,50000,256]
    const int*   ei  = (const int*)d_in[1];           // [2,800000]
    const float* W1  = (const float*)d_in[2];
    const float* b1  = (const float*)d_in[3];
    const float* W2  = (const float*)d_in[4];
    const float* b2  = (const float*)d_in[5];
    float*       out = (float*)d_out;

    const int* src = ei;
    const int* dst = ei + N_EDGES;

    char* ws = (char*)d_ws;
    float*     dinv = (float*)(ws + OFF_DINV);
    _Float16*  Xh   = (_Float16*)(ws + OFF_XH);       // GEMM input (reused layer 2)
    _Float16*  Hg   = (_Float16*)(ws + OFF_HG);       // GEMM output (reused)
    float*     agg  = (float*)(ws + OFF_AGG);         // f32 accumulator (reused)
    _Float16*  W1t  = (_Float16*)(ws + OFF_W1T);
    _Float16*  W2t  = (_Float16*)(ws + OFF_W2T);

    const int NF = N_NODES * DIM;                     // 12,800,000

    // --- normalization -----------------------------------------------------
    deg_init_kernel <<<(N_NODES + 255) / 256, 256, 0, stream>>>(dinv, N_NODES);
    deg_count_kernel<<<(N_EDGES + 255) / 256, 256, 0, stream>>>(dst, dinv, N_EDGES);
    dinv_kernel     <<<(N_NODES + 255) / 256, 256, 0, stream>>>(dinv, N_NODES);

    // --- precision conversion ---------------------------------------------
    f32_to_f16_kernel<<<NF / 256, 256, 0, stream>>>(x, Xh, NF);
    transpose_w_kernel<<<DIM, 256, 0, stream>>>(W1, W1t);
    transpose_w_kernel<<<DIM, 256, 0, stream>>>(W2, W2t);

    const int gemm_blocks = (N_NODES / 16) * 16 / 8;  // 6250 blocks of 8 waves

    // --- layer 1 -----------------------------------------------------------
    gemm_wmma_kernel   <<<gemm_blocks, 256, 0, stream>>>(Xh, W1t, Hg);
    init_agg_kernel    <<<NF / 256, 256, 0, stream>>>(b1, dinv, Hg, agg);
    edge_scatter_kernel<<<(N_EDGES * 32) / 256, 256, 0, stream>>>(src, dst, dinv, Hg, agg);
    relu_f16_kernel    <<<NF / 256, 256, 0, stream>>>(agg, Xh, NF);

    // --- layer 2 -----------------------------------------------------------
    gemm_wmma_kernel   <<<gemm_blocks, 256, 0, stream>>>(Xh, W2t, Hg);
    init_agg_kernel    <<<NF / 256, 256, 0, stream>>>(b2, dinv, Hg, agg);
    edge_scatter_kernel<<<(N_EDGES * 32) / 256, 256, 0, stream>>>(src, dst, dinv, Hg, agg);

    // --- residual epilogue -------------------------------------------------
    finalize_kernel<<<NF / 256, 256, 0, stream>>>(x, agg, out, NF);
}
